// GraphNetBlock_26568667693128
// MI455X (gfx1250) — compile-verified, matrix operand order checked by reading, not run on hardware
//
#include <hip/hip_runtime.h>

typedef __attribute__((ext_vector_type(16))) _Float16 v16h;
typedef __attribute__((ext_vector_type(8)))  _Float16 h8;
typedef __attribute__((ext_vector_type(4)))  _Float16 h4;
typedef __attribute__((ext_vector_type(8)))  float    v8f;

constexpr int kB = 2, kN = 20000, kE = 320000, kH = 128;
constexpr float kEps = 1e-5f;

// ---- strides (in elements) ----
constexpr int E_SA  = 3 * kH + 8;   // 392 halfs, 784B rows (16B aligned)
constexpr int E_SW1 = 3 * kH + 8;   // 392
constexpr int N_SA  = 2 * kH + 8;   // 264
constexpr int N_SW1 = 2 * kH + 8;   // 264
constexpr int S_W2  = kH + 8;       // 136
constexpr int S_H   = kH + 8;       // 136
constexpr int S_O   = kH + 4;       // 132 floats (528B rows, 16B aligned)

// ---- edge-kernel LDS layout (bytes); 64-row tiles ----
constexpr size_t E_OFF_W1   = 0;
constexpr size_t E_OFF_W2   = E_OFF_W1 + (size_t)kH * E_SW1 * 2;  // 100352
constexpr size_t E_OFF_W3   = E_OFF_W2 + (size_t)kH * S_W2 * 2;
constexpr size_t E_OFF_A    = E_OFF_W3 + (size_t)kH * S_W2 * 2;
constexpr size_t E_OFF_H1   = E_OFF_A  + (size_t)64 * E_SA * 2;
constexpr size_t E_OFF_H2   = E_OFF_H1 + (size_t)64 * S_H * 2;
constexpr size_t E_OFF_O    = E_OFF_H2 + (size_t)64 * S_H * 2;
constexpr size_t E_OFF_B1   = E_OFF_O  + (size_t)64 * S_O * 4;
constexpr size_t E_OFF_B2   = E_OFF_B1 + 512;
constexpr size_t E_OFF_B3   = E_OFF_B2 + 512;
constexpr size_t E_OFF_G    = E_OFF_B3 + 512;
constexpr size_t E_OFF_BE   = E_OFF_G  + 512;
constexpr size_t E_OFF_RECV = E_OFF_BE + 512;
constexpr size_t E_OFF_RS   = E_OFF_RECV + 256;
constexpr size_t E_OFF_RQ   = E_OFF_RS + 1024;
constexpr size_t E_OFF_MU   = E_OFF_RQ + 1024;
constexpr size_t E_OFF_SG   = E_OFF_MU + 256;
constexpr size_t E_SMEM     = E_OFF_SG + 256;   // ~287 KB (< 320 KB/WGP)

// ---- node-kernel LDS layout (bytes); 32-row tiles ----
constexpr size_t Nd_OFF_W1 = 0;
constexpr size_t Nd_OFF_W2 = Nd_OFF_W1 + (size_t)kH * N_SW1 * 2;  // 67584
constexpr size_t Nd_OFF_W3 = Nd_OFF_W2 + (size_t)kH * S_W2 * 2;
constexpr size_t Nd_OFF_A  = Nd_OFF_W3 + (size_t)kH * S_W2 * 2;
constexpr size_t Nd_OFF_H1 = Nd_OFF_A  + (size_t)32 * N_SA * 2;
constexpr size_t Nd_OFF_H2 = Nd_OFF_H1 + (size_t)32 * S_H * 2;
constexpr size_t Nd_OFF_O  = Nd_OFF_H2 + (size_t)32 * S_H * 2;
constexpr size_t Nd_OFF_B1 = Nd_OFF_O  + (size_t)32 * S_O * 4;
constexpr size_t Nd_OFF_B2 = Nd_OFF_B1 + 512;
constexpr size_t Nd_OFF_B3 = Nd_OFF_B2 + 512;
constexpr size_t Nd_OFF_G  = Nd_OFF_B3 + 512;
constexpr size_t Nd_OFF_BE = Nd_OFF_G  + 512;
constexpr size_t Nd_OFF_RS = Nd_OFF_BE + 512;
constexpr size_t Nd_OFF_RQ = Nd_OFF_RS + 1024;
constexpr size_t Nd_OFF_MU = Nd_OFF_RQ + 1024;
constexpr size_t Nd_OFF_SG = Nd_OFF_MU + 128;
constexpr size_t Nd_SMEM   = Nd_OFF_SG + 128;   // ~189 KB

// Load a 16-half fragment matching the CDNA5 16-bit A/B VGPR layout:
// lane L holds row/col (L&15); K chunks at (L>>4)*8 and 16+(L>>4)*8.
__device__ __forceinline__ v16h ld_frag(const _Float16* rowptr, int hs) {
  union { v16h v; h8 h[2]; } u;
  u.h[0] = *(const h8*)(rowptr + hs * 8);
  u.h[1] = *(const h8*)(rowptr + 16 + hs * 8);
  return u.v;
}

__device__ __forceinline__ v8f wmma_f16(v16h a, v16h b, v8f c) {
  return __builtin_amdgcn_wmma_f32_16x16x32_f16(false, a, false, b, (short)0, c,
                                                false, false);
}

// 2x2-tile layer for 64-row blocks: each wave owns 2 N-tiles x 2 M-tiles, so
// every A/B fragment is reused twice: 1 LDS fragment load per WMMA.
template <int K, bool RELU, bool OUTF16>
__device__ __forceinline__ void mlp_layer_2x2(const _Float16* sIn, int strideIn,
                                              const _Float16* sWt, int strideW,
                                              const float* bias,
                                              _Float16* outH, float* outF,
                                              int strideOut) {
  const int lane  = threadIdx.x & 31;
  const int wv    = threadIdx.x >> 5;
  const int hs    = lane >> 4;
  const int nl    = lane & 15;
  const int npair = wv & 3;    // which pair of 16-col tiles (0..3)
  const int mpair = wv >> 2;   // which pair of 16-row tiles (0..1)
  const int col0  = npair * 32 + nl;
  const int col1  = col0 + 16;
  const int rbase = mpair * 32;
  const _Float16* arow0 = sIn + (size_t)(rbase + nl) * strideIn;
  const _Float16* arow1 = arow0 + (size_t)16 * strideIn;
  const _Float16* wrow0 = sWt + (size_t)col0 * strideW;
  const _Float16* wrow1 = sWt + (size_t)col1 * strideW;
  v8f acc00 = {}, acc01 = {}, acc10 = {}, acc11 = {};
#pragma unroll
  for (int k0 = 0; k0 < K; k0 += 32) {
    v16h a0 = ld_frag(arow0 + k0, hs);
    v16h a1 = ld_frag(arow1 + k0, hs);
    v16h b0 = ld_frag(wrow0 + k0, hs);
    v16h b1 = ld_frag(wrow1 + k0, hs);
    acc00 = wmma_f16(a0, b0, acc00);
    acc01 = wmma_f16(a0, b1, acc01);
    acc10 = wmma_f16(a1, b0, acc10);
    acc11 = wmma_f16(a1, b1, acc11);
  }
  const float bv0 = bias[col0], bv1 = bias[col1];
#pragma unroll
  for (int r = 0; r < 8; ++r) {
    const int rl = r + hs * 8;
    const int m0 = rbase + rl, m1 = rbase + 16 + rl;
    float v00 = acc00[r] + bv0, v01 = acc01[r] + bv1;
    float v10 = acc10[r] + bv0, v11 = acc11[r] + bv1;
    if (RELU) {
      v00 = fmaxf(v00, 0.f); v01 = fmaxf(v01, 0.f);
      v10 = fmaxf(v10, 0.f); v11 = fmaxf(v11, 0.f);
    }
    if (OUTF16) {
      outH[(size_t)m0 * strideOut + col0] = (_Float16)v00;
      outH[(size_t)m0 * strideOut + col1] = (_Float16)v01;
      outH[(size_t)m1 * strideOut + col0] = (_Float16)v10;
      outH[(size_t)m1 * strideOut + col1] = (_Float16)v11;
    } else {
      outF[(size_t)m0 * strideOut + col0] = v00;
      outF[(size_t)m0 * strideOut + col1] = v01;
      outF[(size_t)m1 * strideOut + col0] = v10;
      outF[(size_t)m1 * strideOut + col1] = v11;
    }
  }
}

// 1x2-tile layer for 32-row blocks (node kernel): wave owns 1 N-tile, 2 M-tiles.
template <int K, bool RELU, bool OUTF16>
__device__ __forceinline__ void mlp_layer_1x2(const _Float16* sIn, int strideIn,
                                              const _Float16* sWt, int strideW,
                                              const float* bias,
                                              _Float16* outH, float* outF,
                                              int strideOut) {
  const int lane = threadIdx.x & 31;
  const int wv   = threadIdx.x >> 5;
  const int hs   = lane >> 4;
  const int nl   = lane & 15;
  const int colg = wv * 16 + nl;
  const _Float16* wrow  = sWt + (size_t)colg * strideW;
  const _Float16* arow0 = sIn + (size_t)nl * strideIn;
  const _Float16* arow1 = sIn + (size_t)(nl + 16) * strideIn;
  v8f acc0 = {}, acc1 = {};
#pragma unroll
  for (int k0 = 0; k0 < K; k0 += 32) {
    v16h bf = ld_frag(wrow + k0, hs);
    v16h a0 = ld_frag(arow0 + k0, hs);
    acc0 = wmma_f16(a0, bf, acc0);
    v16h a1 = ld_frag(arow1 + k0, hs);
    acc1 = wmma_f16(a1, bf, acc1);
  }
  const float bv = bias[colg];
#pragma unroll
  for (int r = 0; r < 8; ++r) {
    const int rl = r + hs * 8;
    float v0 = acc0[r] + bv, v1 = acc1[r] + bv;
    if (RELU) { v0 = fmaxf(v0, 0.f); v1 = fmaxf(v1, 0.f); }
    if (OUTF16) {
      outH[(size_t)rl * strideOut + colg]        = (_Float16)v0;
      outH[(size_t)(rl + 16) * strideOut + colg] = (_Float16)v1;
    } else {
      outF[(size_t)rl * strideOut + colg]        = v0;
      outF[(size_t)(rl + 16) * strideOut + colg] = v1;
    }
  }
}

__global__ void zero_kernel(float4* p, int n4) {
  int i = blockIdx.x * 256 + threadIdx.x;
  if (i < n4) p[i] = make_float4(0.f, 0.f, 0.f, 0.f);
}

// Edge update: gather endpoints, concat, 3-layer MLP + LN, residual write,
// fused scatter-sum into accbuf via native f32 atomics. 64 edges per iteration.
__global__ __launch_bounds__(256) void edge_kernel(
    const int* __restrict__ senders, const int* __restrict__ receivers,
    const float* __restrict__ ef, const float* __restrict__ snf,
    const float* __restrict__ rnf,
    const float* __restrict__ w1, const float* __restrict__ b1,
    const float* __restrict__ w2, const float* __restrict__ b2,
    const float* __restrict__ w3, const float* __restrict__ b3,
    const float* __restrict__ g, const float* __restrict__ be,
    float* __restrict__ out_edge, float* __restrict__ accbuf) {
  extern __shared__ char smem[];
  _Float16* sW1t = (_Float16*)(smem + E_OFF_W1);
  _Float16* sW2t = (_Float16*)(smem + E_OFF_W2);
  _Float16* sW3t = (_Float16*)(smem + E_OFF_W3);
  _Float16* sA   = (_Float16*)(smem + E_OFF_A);
  _Float16* sH1  = (_Float16*)(smem + E_OFF_H1);
  _Float16* sH2  = (_Float16*)(smem + E_OFF_H2);
  float* sOut  = (float*)(smem + E_OFF_O);
  float* sB1   = (float*)(smem + E_OFF_B1);
  float* sB2   = (float*)(smem + E_OFF_B2);
  float* sB3   = (float*)(smem + E_OFF_B3);
  float* sG    = (float*)(smem + E_OFF_G);
  float* sBe   = (float*)(smem + E_OFF_BE);
  int*   sRecv = (int*)(smem + E_OFF_RECV);
  float* sRedS = (float*)(smem + E_OFF_RS);
  float* sRedQ = (float*)(smem + E_OFF_RQ);
  float* sMu   = (float*)(smem + E_OFF_MU);
  float* sSg   = (float*)(smem + E_OFF_SG);

  const int tid = threadIdx.x;
  // Persistent weights: transpose to Wt[n][k] in f16 once per block.
  for (int i = tid; i < 3 * kH * kH; i += 256) {
    int k = i >> 7, n = i & 127;
    sW1t[(size_t)n * E_SW1 + k] = (_Float16)w1[i];
  }
  for (int i = tid; i < kH * kH; i += 256) {
    int k = i >> 7, n = i & 127;
    sW2t[(size_t)n * S_W2 + k] = (_Float16)w2[i];
    sW3t[(size_t)n * S_W2 + k] = (_Float16)w3[i];
  }
  for (int i = tid; i < kH; i += 256) {
    sB1[i] = b1[i]; sB2[i] = b2[i]; sB3[i] = b3[i];
    sG[i] = g[i];   sBe[i] = be[i];
  }
  __syncthreads();

  const int lane = tid & 31, grp = tid >> 5;
  const int tilesPerB = kE / 64;  // 5000
  for (int t = blockIdx.x; t < kB * tilesPerB; t += gridDim.x) {
    const int b  = t / tilesPerB;
    const int e0 = (t - b * tilesPerB) * 64;
    __syncthreads();  // protect sA/sRecv/sOut reuse across iterations
    if (tid < 64) sRecv[tid] = receivers[(size_t)b * kE + e0 + tid];
    // Fill A = [sender_feat | receiver_feat | edge_feat], f32 -> f16.
    for (int sr = grp; sr < 192; sr += 8) {
      const int row = sr / 3, seg = sr - row * 3;
      const float* src;
      if (seg == 0)
        src = snf + ((size_t)b * kN + senders[(size_t)b * kE + e0 + row]) * kH;
      else if (seg == 1)
        src = rnf + ((size_t)b * kN + receivers[(size_t)b * kE + e0 + row]) * kH;
      else
        src = ef + ((size_t)b * kE + e0 + row) * kH;
      float4 v = ((const float4*)src)[lane];
      h4 hv = {(_Float16)v.x, (_Float16)v.y, (_Float16)v.z, (_Float16)v.w};
      *(h4*)(sA + (size_t)row * E_SA + seg * kH + lane * 4) = hv;
    }
    __syncthreads();
    mlp_layer_2x2<3 * kH, true, true>(sA, E_SA, sW1t, E_SW1, sB1, sH1, nullptr, S_H);
    __syncthreads();
    mlp_layer_2x2<kH, true, true>(sH1, S_H, sW2t, S_W2, sB2, sH2, nullptr, S_H);
    __syncthreads();
    mlp_layer_2x2<kH, false, false>(sH2, S_H, sW3t, S_W2, sB3, nullptr, sOut, S_O);
    __syncthreads();
    // LayerNorm over H=128 per row (4 threads/row x 8 float4s each).
    {
      const int row = tid >> 2, sub = tid & 3;
      const float4* rp = (const float4*)(sOut + (size_t)row * S_O);
      float s = 0.f, q = 0.f;
#pragma unroll
      for (int j = 0; j < 8; ++j) {
        float4 x = rp[sub + 4 * j];
        s += x.x + x.y + x.z + x.w;
        q += x.x * x.x + x.y * x.y + x.z * x.z + x.w * x.w;
      }
      sRedS[row * 4 + sub] = s; sRedQ[row * 4 + sub] = q;
    }
    __syncthreads();
    if ((tid & 3) == 0) {
      const int row = tid >> 2;
      float s = 0.f, q = 0.f;
#pragma unroll
      for (int i = 0; i < 4; ++i) { s += sRedS[row * 4 + i]; q += sRedQ[row * 4 + i]; }
      const float mu = s * (1.0f / kH);
      const float var = q * (1.0f / kH) - mu * mu;
      sMu[row] = mu; sSg[row] = rsqrtf(var + kEps);
    }
    __syncthreads();
    // Apply LN, write residual output (float4, coalesced), scatter-sum.
    {
      const int row = tid >> 2, sub = tid & 3;
      const int e = e0 + row, recv = sRecv[row];
      const float mu = sMu[row], rs = sSg[row];
      const float4* hrow = (const float4*)(sOut + (size_t)row * S_O);
      const float4* gv4  = (const float4*)sG;
      const float4* be4  = (const float4*)sBe;
      const float4* efr  = (const float4*)(ef + ((size_t)b * kE + e) * kH);
      float4* orow = (float4*)(out_edge + ((size_t)b * kE + e) * kH);
      float* arow  = accbuf + ((size_t)b * kN + recv) * kH;
#pragma unroll
      for (int j = 0; j < 8; ++j) {
        const int f4 = sub + 4 * j;
        float4 h = hrow[f4], gg = gv4[f4], bb = be4[f4], rr = efr[f4];
        float4 v;
        v.x = (h.x - mu) * rs * gg.x + bb.x;
        v.y = (h.y - mu) * rs * gg.y + bb.y;
        v.z = (h.z - mu) * rs * gg.z + bb.z;
        v.w = (h.w - mu) * rs * gg.w + bb.w;
        orow[f4] = make_float4(v.x + rr.x, v.y + rr.y, v.z + rr.z, v.w + rr.w);
        unsafeAtomicAdd(arow + 4 * f4 + 0, v.x);
        unsafeAtomicAdd(arow + 4 * f4 + 1, v.y);
        unsafeAtomicAdd(arow + 4 * f4 + 2, v.z);
        unsafeAtomicAdd(arow + 4 * f4 + 3, v.w);
      }
    }
  }
}

// Node update: concat [node_feat | acc], 3-layer MLP + LN, residual write.
__global__ __launch_bounds__(256) void node_kernel(
    const float* __restrict__ rnf, const float* __restrict__ accbuf,
    const float* __restrict__ w1, const float* __restrict__ b1,
    const float* __restrict__ w2, const float* __restrict__ b2,
    const float* __restrict__ w3, const float* __restrict__ b3,
    const float* __restrict__ g, const float* __restrict__ be,
    float* __restrict__ out_node) {
  extern __shared__ char smem[];
  _Float16* sW1t = (_Float16*)(smem + Nd_OFF_W1);
  _Float16* sW2t = (_Float16*)(smem + Nd_OFF_W2);
  _Float16* sW3t = (_Float16*)(smem + Nd_OFF_W3);
  _Float16* sA   = (_Float16*)(smem + Nd_OFF_A);
  _Float16* sH1  = (_Float16*)(smem + Nd_OFF_H1);
  _Float16* sH2  = (_Float16*)(smem + Nd_OFF_H2);
  float* sOut  = (float*)(smem + Nd_OFF_O);
  float* sB1   = (float*)(smem + Nd_OFF_B1);
  float* sB2   = (float*)(smem + Nd_OFF_B2);
  float* sB3   = (float*)(smem + Nd_OFF_B3);
  float* sG    = (float*)(smem + Nd_OFF_G);
  float* sBe   = (float*)(smem + Nd_OFF_BE);
  float* sRedS = (float*)(smem + Nd_OFF_RS);
  float* sRedQ = (float*)(smem + Nd_OFF_RQ);
  float* sMu   = (float*)(smem + Nd_OFF_MU);
  float* sSg   = (float*)(smem + Nd_OFF_SG);

  const int tid = threadIdx.x;
  for (int i = tid; i < 2 * kH * kH; i += 256) {
    int k = i >> 7, n = i & 127;
    sW1t[(size_t)n * N_SW1 + k] = (_Float16)w1[i];
  }
  for (int i = tid; i < kH * kH; i += 256) {
    int k = i >> 7, n = i & 127;
    sW2t[(size_t)n * S_W2 + k] = (_Float16)w2[i];
    sW3t[(size_t)n * S_W2 + k] = (_Float16)w3[i];
  }
  for (int i = tid; i < kH; i += 256) {
    sB1[i] = b1[i]; sB2[i] = b2[i]; sB3[i] = b3[i];
    sG[i] = g[i];   sBe[i] = be[i];
  }
  __syncthreads();

  const int lane = tid & 31, grp = tid >> 5;
  const int tilesPerB = kN / 32;  // 625
  for (int t = blockIdx.x; t < kB * tilesPerB; t += gridDim.x) {
    const int b  = t / tilesPerB;
    const int n0 = (t - b * tilesPerB) * 32;
    __syncthreads();
    for (int sr = grp; sr < 64; sr += 8) {
      const int row = sr >> 1, seg = sr & 1;
      const float* src = (seg == 0)
          ? rnf + ((size_t)b * kN + n0 + row) * kH
          : accbuf + ((size_t)b * kN + n0 + row) * kH;
      float4 v = ((const float4*)src)[lane];
      h4 hv = {(_Float16)v.x, (_Float16)v.y, (_Float16)v.z, (_Float16)v.w};
      *(h4*)(sA + (size_t)row * N_SA + seg * kH + lane * 4) = hv;
    }
    __syncthreads();
    mlp_layer_1x2<2 * kH, true, true>(sA, N_SA, sW1t, N_SW1, sB1, sH1, nullptr, S_H);
    __syncthreads();
    mlp_layer_1x2<kH, true, true>(sH1, S_H, sW2t, S_W2, sB2, sH2, nullptr, S_H);
    __syncthreads();
    mlp_layer_1x2<kH, false, false>(sH2, S_H, sW3t, S_W2, sB3, nullptr, sOut, S_O);
    __syncthreads();
    {
      const int row = tid >> 3, sub = tid & 7;
      const float4* rp = (const float4*)(sOut + (size_t)row * S_O);
      float s = 0.f, q = 0.f;
#pragma unroll
      for (int j = 0; j < 4; ++j) {
        float4 x = rp[sub + 8 * j];
        s += x.x + x.y + x.z + x.w;
        q += x.x * x.x + x.y * x.y + x.z * x.z + x.w * x.w;
      }
      sRedS[row * 8 + sub] = s; sRedQ[row * 8 + sub] = q;
    }
    __syncthreads();
    if ((tid & 7) == 0) {
      const int row = tid >> 3;
      float s = 0.f, q = 0.f;
#pragma unroll
      for (int i = 0; i < 8; ++i) { s += sRedS[row * 8 + i]; q += sRedQ[row * 8 + i]; }
      const float mu = s * (1.0f / kH);
      const float var = q * (1.0f / kH) - mu * mu;
      sMu[row] = mu; sSg[row] = rsqrtf(var + kEps);
    }
    __syncthreads();
    {
      const int row = tid >> 3, sub = tid & 7;
      const int n = n0 + row;
      const float mu = sMu[row], rs = sSg[row];
      const float4* hrow = (const float4*)(sOut + (size_t)row * S_O);
      const float4* gv4  = (const float4*)sG;
      const float4* be4  = (const float4*)sBe;
      const float4* nfr  = (const float4*)(rnf + ((size_t)b * kN + n) * kH);
      float4* orow = (float4*)(out_node + ((size_t)b * kN + n) * kH);
#pragma unroll
      for (int j = 0; j < 4; ++j) {
        const int f4 = sub + 8 * j;
        float4 h = hrow[f4], gg = gv4[f4], bb = be4[f4], rr = nfr[f4];
        float4 o;
        o.x = (h.x - mu) * rs * gg.x + bb.x + rr.x;
        o.y = (h.y - mu) * rs * gg.y + bb.y + rr.y;
        o.z = (h.z - mu) * rs * gg.z + bb.z + rr.z;
        o.w = (h.w - mu) * rs * gg.w + bb.w + rr.w;
        orow[f4] = o;
      }
    }
  }
}

extern "C" void kernel_launch(void* const* d_in, const int* in_sizes, int n_in,
                              void* d_out, int out_size, void* d_ws,
                              size_t ws_size, hipStream_t stream) {
  (void)in_sizes; (void)n_in; (void)out_size; (void)ws_size;
  const int*   senders   = (const int*)d_in[0];
  const int*   receivers = (const int*)d_in[1];
  const float* ef  = (const float*)d_in[2];
  const float* snf = (const float*)d_in[3];
  const float* rnf = (const float*)d_in[4];
  const float* ew1 = (const float*)d_in[5];
  const float* eb1 = (const float*)d_in[6];
  const float* ew2 = (const float*)d_in[7];
  const float* eb2 = (const float*)d_in[8];
  const float* ew3 = (const float*)d_in[9];
  const float* eb3 = (const float*)d_in[10];
  const float* eg  = (const float*)d_in[11];
  const float* ebe = (const float*)d_in[12];
  const float* nw1 = (const float*)d_in[13];
  const float* nb1 = (const float*)d_in[14];
  const float* nw2 = (const float*)d_in[15];
  const float* nb2 = (const float*)d_in[16];
  const float* nw3 = (const float*)d_in[17];
  const float* nb3 = (const float*)d_in[18];
  const float* ng  = (const float*)d_in[19];
  const float* nbe = (const float*)d_in[20];

  float* out_edge = (float*)d_out;
  float* out_node = out_edge + (size_t)kB * kE * kH;
  float* accbuf   = (float*)d_ws;  // [B,N,H] f32 scatter accumulator (20.5 MB)

  const int n4 = (kB * kN * kH) / 4;
  zero_kernel<<<(n4 + 255) / 256, 256, 0, stream>>>((float4*)accbuf, n4);
  edge_kernel<<<512, 256, (unsigned)E_SMEM, stream>>>(
      senders, receivers, ef, snf, rnf, ew1, eb1, ew2, eb2, ew3, eb3, eg, ebe,
      out_edge, accbuf);
  node_kernel<<<256, 256, (unsigned)Nd_SMEM, stream>>>(
      rnf, accbuf, nw1, nb1, nw2, nb2, nw3, nb3, ng, nbe, out_node);
}
